// FERNN_Cell_w_VChannels_46806553592095
// MI455X (gfx1250) — matrix operand order repaired and checked by reading, not compiled, not run on hardware
//
#include <hip/hip_runtime.h>

typedef __bf16 bf16_t;
typedef __attribute__((ext_vector_type(16))) __bf16 v16bf;
typedef __attribute__((ext_vector_type(8)))  float  v8f;
typedef __attribute__((ext_vector_type(4)))  unsigned int u32x4;

union FragBF {
    v16bf v;
    u32x4 q[2];
};

#define CI_PAD 40   // 32 channels padded to 40 bf16 (80B) -> 16B-aligned vector loads

// ---------------------------------------------------------------------------
// Prep 1: convert W_h [128co][128ci][3][3] f32 -> bf16 workspace [tap][co][ci]
// ---------------------------------------------------------------------------
__global__ __launch_bounds__(256) void wprep_kernel(const float* __restrict__ W_h,
                                                    bf16_t* __restrict__ w_bf) {
    int idx = blockIdx.x * 256 + threadIdx.x;
    if (idx >= 9 * 128 * 128) return;
    int ci  = idx & 127;
    int co  = (idx >> 7) & 127;
    int tap = idx >> 14;
    w_bf[idx] = (bf16_t)W_h[(co * 128 + ci) * 9 + tap];
}

// ---------------------------------------------------------------------------
// Prep 2: encoder circular conv u_conv[b][co][y][x] = conv(u_t, W_u), 32x32
// (0.1 GFLOP -> plain VALU is fine)
// ---------------------------------------------------------------------------
__global__ __launch_bounds__(256) void uconv_kernel(const float* __restrict__ u_t,
                                                    const float* __restrict__ W_u,
                                                    float* __restrict__ uc) {
    int idx = blockIdx.x * 256 + threadIdx.x;   // 16*128*32*32 = 2097152
    int x  = idx & 31;
    int y  = (idx >> 5) & 31;
    int co = (idx >> 10) & 127;
    int b  = idx >> 17;
    float s = 0.f;
#pragma unroll
    for (int ci = 0; ci < 3; ++ci)
#pragma unroll
        for (int ky = 0; ky < 3; ++ky)
#pragma unroll
            for (int kx = 0; kx < 3; ++kx)
                s += W_u[((co * 3 + ci) * 3 + ky) * 3 + kx] *
                     u_t[((b * 3 + ci) * 32 + ((y + ky - 1) & 31)) * 32 + ((x + kx - 1) & 31)];
    uc[idx] = s;
}

// ---------------------------------------------------------------------------
// Main: fused roll + recurrent conv (implicit GEMM via bf16 WMMA) + u + ReLU
// Grid: x = output row y (64), y = velocity v (9), z = batch b (16)
// Block: 256 threads = 8 waves; wave w -> Cout rows [16w,16w+16) x 64 pixels
// ---------------------------------------------------------------------------
__global__ __launch_bounds__(256) void fernn_main_kernel(
    const float* __restrict__ h_prev, const int* __restrict__ action,
    const float* __restrict__ u_conv, const bf16_t* __restrict__ w_bf,
    float* __restrict__ out) {

    __shared__ bf16_t lds[3 * 64 * CI_PAD];   // 3 circular rows x 64 x 32 ch (padded)

    const int y  = blockIdx.x;
    const int v  = blockIdx.y;
    const int b  = blockIdx.z;
    const int vx = (v / 3) - 1;               // V_LIST order: vx outer, vy inner
    const int vy = (v % 3) - 1;
    const int ax = action[2 * b + 0];
    const int ay = action[2 * b + 1];

    const int tid   = threadIdx.x;
    const int lane  = tid & 31;
    const int wave  = tid >> 5;
    const int co0   = wave * 16;
    const int m     = lane & 15;              // M row (A) / N column (B)
    const int khalf = lane >> 4;              // lane half selects K region

    const float* hb = h_prev + (size_t)(b * 9 + v) * (128 * 64 * 64);

    v8f acc[4] = {};                          // 4 pixel groups of 16 -> 64 pixels

    for (int chunk = 0; chunk < 4; ++chunk) { // Cin in chunks of 32
        if (chunk) __syncthreads();
        // Stage h_shifted rows y-1,y,y+1 (all rolls folded into indices).
        // Two consecutive channels per thread -> one packed ds_store_b32;
        // global reads stay coalesced along x.
#pragma unroll
        for (int it = 0; it < 12; ++it) {     // 3*64*16 pairs / 256 threads
            int f  = tid + it * 256;
            int x  = f & 63;
            int cp = (f >> 6) & 15;           // channel pair index
            int r  = f >> 10;                 // 0..2 (tap row)
            int ci = cp * 2;
            int csrc0 = (chunk * 32 + ci     - vy) & 127;   // channel roll by vy
            int csrc1 = (chunk * 32 + ci + 1 - vy) & 127;
            int ysrc  = (y + r - 1 + ay - vx) & 63;         // H roll by vx, action ay
            int xsrc  = (x + ax) & 63;                      // W roll by action ax
            size_t sp = (size_t)ysrc * 64 + xsrc;
            union { bf16_t h[2]; unsigned int u; } pk;
            pk.h[0] = (bf16_t)hb[(size_t)csrc0 * 4096 + sp];
            pk.h[1] = (bf16_t)hb[(size_t)csrc1 * 4096 + sp];
            *(unsigned int*)&lds[(r * 64 + x) * CI_PAD + ci] = pk.u;
        }
        __syncthreads();

#pragma unroll
        for (int tap = 0; tap < 9; ++tap) {
            const int ty = tap / 3;           // LDS row
            const int tx = tap % 3;           // x offset (dx = tx-1)
            // A fragment: 16x32 bf16 weights, documented lane layout:
            // lane half 0 -> K 0..7 & 16..23, half 1 -> K 8..15 & 24..31
            FragBF a;
            const bf16_t* wp = w_bf + ((tap * 128 + (co0 + m)) * 128 + chunk * 32 + khalf * 8);
            a.q[0] = *(const u32x4*)(wp);
            a.q[1] = *(const u32x4*)(wp + 16);
            // Load ALL 4 B fragments into distinct registers first so the
            // ds_loads pipeline instead of each WMMA waiting on dscnt==0.
            FragBF bb[4];
#pragma unroll
            for (int g = 0; g < 4; ++g) {
                int xx = (g * 16 + m + tx - 1) & 63;        // circular x
                const bf16_t* bp = &lds[(ty * 64 + xx) * CI_PAD + khalf * 8];
                bb[g].q[0] = *(const u32x4*)(bp);
                bb[g].q[1] = *(const u32x4*)(bp + 16);
            }
#pragma unroll
            for (int g = 0; g < 4; ++g)
                acc[g] = __builtin_amdgcn_wmma_f32_16x16x32_bf16(
                    false, a.v, false, bb[g].v, (short)0, acc[g], false, false);
        }
    }

    // Epilogue: + encoder window, ReLU, store (C/D layout: VGPR r -> M, lane -> N)
    float* ob = out + (size_t)((b * 9 + v) * 128) * 4096;
#pragma unroll
    for (int g = 0; g < 4; ++g) {
        int x = g * 16 + m;
#pragma unroll
        for (int r = 0; r < 8; ++r) {
            int co = co0 + r + khalf * 8;
            float s = acc[g][r];
            if (y < 32 && x < 32)
                s += u_conv[((b * 128 + co) * 32 + y) * 32 + x];
            ob[(size_t)co * 4096 + y * 64 + x] = s > 0.f ? s : 0.f;
        }
    }
}

// ---------------------------------------------------------------------------
extern "C" void kernel_launch(void* const* d_in, const int* in_sizes, int n_in,
                              void* d_out, int out_size, void* d_ws, size_t ws_size,
                              hipStream_t stream) {
    const float* u_t    = (const float*)d_in[0];   // [16,3,32,32]
    const float* h_prev = (const float*)d_in[1];   // [16,9,128,64,64]
    const int*   action = (const int*)d_in[2];     // [16,2]
    const float* W_u    = (const float*)d_in[3];   // [128,3,3,3]
    const float* W_h    = (const float*)d_in[4];   // [128,128,3,3]
    float* out = (float*)d_out;                    // [16,9,128,64,64]

    float*  uc  = (float*)d_ws;                                            // 8 MB
    bf16_t* wbf = (bf16_t*)((char*)d_ws + (size_t)16 * 128 * 32 * 32 * 4); // 288 KB

    uconv_kernel<<<(16 * 128 * 32 * 32) / 256, 256, 0, stream>>>(u_t, W_u, uc);
    wprep_kernel<<<(9 * 128 * 128 + 255) / 256, 256, 0, stream>>>(W_h, wbf);

    dim3 grid(64, 9, 16);
    fernn_main_kernel<<<grid, 256, 0, stream>>>(h_prev, action, uc, wbf, out);
}
